// StemStage_3058016715337
// MI455X (gfx1250) — matrix operand order repaired and checked by reading, not compiled, not run on hardware
//
#include <hip/hip_runtime.h>

typedef __attribute__((ext_vector_type(16))) _Float16 v16h;
typedef __attribute__((ext_vector_type(8)))  float    v8f;

#define N_PTS 131072
#define KNBR  27
#define FOUT  32
#define EPS_F 1e-5f

// Workspace layout (bytes):
//   [0,        55296)  : w2 packed f16 in WMMA-B lane layout (27*2*32*16 halves)
//   [55296,    55808)  : folded BN constants (4 x 32 f32: s1, b1, s2, b2)
//   [57344, 57344+8MB) : h16, N*32 f16 (stage-1 activations)
#define W2P_OFF   0
#define CONST_OFF 55296
#define H16_OFF   57344

// ---------------------------------------------------------------------------
// Kernel 0: pack w2 -> f16 in the per-lane B-operand layout; fold BN consts.
// B layout used here (contiguous K, same order as A): lane l holds column n = l&15;
// element e of its v16h is K = (l>>4)*16 + e.
// ---------------------------------------------------------------------------
__global__ __launch_bounds__(256) void pack_kernel(
    const float* __restrict__ w2,
    const float* __restrict__ g1, const float* __restrict__ b1,
    const float* __restrict__ m1, const float* __restrict__ v1,
    const float* __restrict__ mg, const float* __restrict__ mbeta,
    const float* __restrict__ mm, const float* __restrict__ mv,
    _Float16* __restrict__ w2p, float* __restrict__ consts) {
  int j = blockIdx.x * blockDim.x + threadIdx.x;
  if (j < KNBR * 2 * 32 * 16) {
    int e = j & 15;
    int l = (j >> 4) & 31;
    int t = (j >> 9) & 1;
    int k = j >> 10;
    int kk = ((l >> 4) << 4) + e;        // K index this lane element holds
    int o  = (t << 4) + (l & 15);        // output column
    w2p[j] = (_Float16)w2[(k * 32 + kk) * 32 + o];
  }
  if (j < 32) {
    float s1 = g1[j] * rsqrtf(v1[j] + EPS_F);
    consts[j]      = s1;
    consts[32 + j] = b1[j] - m1[j] * s1;
    float s2 = mg[j] * rsqrtf(mv[j] + EPS_F);
    consts[64 + j] = s2;
    consts[96 + j] = mbeta[j] - mm[j] * s2;
  }
}

// ---------------------------------------------------------------------------
// Kernel 1: stage-1 gather-GEMM (K=3 -> plain f32 FMA), BN+SiLU -> h16,
// plus the point branch (Linear+BN+ReLU) written straight into d_out copy 0.
// Mapping: one wave32 per point, lane = output channel. w1 cached in LDS.
// ---------------------------------------------------------------------------
__global__ __launch_bounds__(256) void stage1_kernel(
    const float* __restrict__ x, const float* __restrict__ z,
    const int* __restrict__ nbr, const float* __restrict__ w1,
    const float* __restrict__ mlp_w, const float* __restrict__ mlp_b,
    const float* __restrict__ consts,
    _Float16* __restrict__ h16, float* __restrict__ out) {
  __shared__ float w1s[KNBR * 3 * 32];
  for (int i = threadIdx.x; i < KNBR * 3 * 32; i += 256) w1s[i] = w1[i];
  __syncthreads();

  const int lane = threadIdx.x & 31;
  const int n = blockIdx.x * 8 + (threadIdx.x >> 5);
  const int* __restrict__ idxp = nbr + n * KNBR;

  float acc = 0.f;
#pragma unroll 3
  for (int k = 0; k < KNBR; ++k) {
    int idx = idxp[k];                       // uniform across wave
    float x0 = x[idx * 3 + 0];
    float x1 = x[idx * 3 + 1];
    float x2 = x[idx * 3 + 2];
    const float* wk = w1s + k * 96 + lane;   // per-lane channel, bank-clean
    acc = fmaf(x0, wk[0],  acc);
    acc = fmaf(x1, wk[32], acc);
    acc = fmaf(x2, wk[64], acc);
  }
  // BN + SiLU
  float hb = fmaf(acc, consts[lane], consts[32 + lane]);
  float h  = hb / (1.f + __expf(-hb));
  h16[n * 32 + lane] = (_Float16)h;

  // point branch: z @ mlp_w + b -> BN -> ReLU, staged into d_out copy 0
  float z0 = z[n * 3 + 0], z1 = z[n * 3 + 1], z2 = z[n * 3 + 2];
  float za = mlp_b[lane];
  za = fmaf(z0, mlp_w[lane],      za);
  za = fmaf(z1, mlp_w[32 + lane], za);
  za = fmaf(z2, mlp_w[64 + lane], za);
  float zb = fmaf(za, consts[64 + lane], consts[96 + lane]);
  out[n * 32 + lane] = fmaxf(zb, 0.f);
}

// ---------------------------------------------------------------------------
// Kernel 2: stage-2 gather-GEMM on WMMA. One wave per 16-point tile.
// Per k: A = 16x32 f16 gathered rows of h16 (A layout matching B: lane l holds
// row l&15; elem e -> K=(l>>4)*16+e, i.e. one contiguous 32-byte span per
// lane), B from LDS, two f32-accum WMMAs.
// ---------------------------------------------------------------------------
__global__ __launch_bounds__(256) void stage2_wmma_kernel(
    const int* __restrict__ nbr, const _Float16* __restrict__ h16raw,
    const _Float16* __restrict__ w2p, float* __restrict__ out) {
  __shared__ v16h w2s[KNBR * 2 * 32];        // 55296 B
  {
    const uint4* __restrict__ src = (const uint4*)w2p;
    uint4* dst = (uint4*)w2s;
    for (int i = threadIdx.x; i < KNBR * 2 * 32 * 2; i += 256) dst[i] = src[i];
  }
  __syncthreads();

  const int lane = threadIdx.x & 31;
  const int hv   = lane >> 4;                // wave half
  const int row  = lane & 15;
  const int tile = blockIdx.x * 8 + (threadIdx.x >> 5);
  const int nbase = tile * 16;
  const int* __restrict__ idxp = nbr + (nbase + row) * KNBR;
  const char* __restrict__ hbase = (const char*)h16raw;

  v8f acc0 = {};
  v8f acc1 = {};
  for (int k = 0; k < KNBR; ++k) {
    int idx = idxp[k];
    const char* hr = hbase + idx * 64 + hv * 32;   // 64 B per h row
    union { uint4 q[2]; v16h v; } a;
    a.q[0] = *(const uint4*)(hr);                  // K chunk 0
    a.q[1] = *(const uint4*)(hr + 16);             // K chunk 1
    v16h b0 = w2s[(k * 2 + 0) * 32 + lane];
    v16h b1 = w2s[(k * 2 + 1) * 32 + lane];
    acc0 = __builtin_amdgcn_wmma_f32_16x16x32_f16(
        false, a.v, false, b0, (short)0, acc0, false, false);
    acc1 = __builtin_amdgcn_wmma_f32_16x16x32_f16(
        false, a.v, false, b1, (short)0, acc1, false, false);
  }

  // C layout: VGPR r, lane l -> point nbase + r + 8*(l>>4), col (l&15) + 16t.
  // Fuse with point branch (already in out copy 0), write both tuple copies.
  const int col = lane & 15;
#pragma unroll
  for (int r = 0; r < 8; ++r) {
    int n = nbase + r + 8 * hv;
    float* p0 = out + n * 32 + col;
    float v0 = acc0[r] + p0[0];
    p0[0] = v0;
    out[N_PTS * 32 + n * 32 + col] = v0;
    float* p1 = out + n * 32 + 16 + col;
    float v1 = acc1[r] + p1[0];
    p1[0] = v1;
    out[N_PTS * 32 + n * 32 + 16 + col] = v1;
  }
}

// ---------------------------------------------------------------------------
extern "C" void kernel_launch(void* const* d_in, const int* in_sizes, int n_in,
                              void* d_out, int out_size, void* d_ws, size_t ws_size,
                              hipStream_t stream) {
  const float* x     = (const float*)d_in[0];
  const float* z     = (const float*)d_in[1];
  const int*   nbr   = (const int*)  d_in[2];
  const float* w1    = (const float*)d_in[3];
  const float* bn1g  = (const float*)d_in[4];
  const float* bn1b  = (const float*)d_in[5];
  const float* bn1m  = (const float*)d_in[6];
  const float* bn1v  = (const float*)d_in[7];
  const float* w2    = (const float*)d_in[8];
  const float* mlpw  = (const float*)d_in[9];
  const float* mlpb  = (const float*)d_in[10];
  const float* mlpg  = (const float*)d_in[11];
  const float* mlpbt = (const float*)d_in[12];
  const float* mlpm  = (const float*)d_in[13];
  const float* mlpv  = (const float*)d_in[14];

  float* out = (float*)d_out;
  char*  ws  = (char*)d_ws;
  _Float16* w2p    = (_Float16*)(ws + W2P_OFF);
  float*    consts = (float*)   (ws + CONST_OFF);
  _Float16* h16    = (_Float16*)(ws + H16_OFF);

  pack_kernel<<<108, 256, 0, stream>>>(w2, bn1g, bn1b, bn1m, bn1v,
                                       mlpg, mlpbt, mlpm, mlpv, w2p, consts);
  stage1_kernel<<<N_PTS / 8, 256, 0, stream>>>(x, z, nbr, w1, mlpw, mlpb,
                                               consts, h16, out);
  stage2_wmma_kernel<<<N_PTS / 16 / 8, 256, 0, stream>>>(nbr, h16, w2p, out);
}
